// RotaryMultiheadAttention_17738214933020
// MI455X (gfx1250) — matrix operand order repaired and checked by reading, not compiled, not run on hardware
//
#include <hip/hip_runtime.h>

typedef __attribute__((ext_vector_type(16))) _Float16 v16h;
typedef __attribute__((ext_vector_type(8)))  _Float16 v8h;
typedef __attribute__((ext_vector_type(8)))  float    v8f;

#define DIM   256
#define NSEQ  2048
#define NB    8
#define MTOT  (NB * NSEQ)               // 16384 rows
#define SCALE 0.17677669529663687f     // 1/sqrt(32)

// Async memory->LDS path (ASYNCcnt): builtin exists on this toolchain with
// signature (v4i addrspace(1)*, v4i addrspace(3)*, imm offset, imm cpol).
// Materialize the numbered-AS pointers via integer casts (always legal).
#if defined(__gfx1250__) && __has_builtin(__builtin_amdgcn_global_load_async_to_lds_b128)
#define HAVE_ASYNC_LDS 1
typedef __attribute__((ext_vector_type(4))) int v4i;
typedef __attribute__((address_space(1))) v4i g_v4i;
typedef __attribute__((address_space(3))) v4i l_v4i;
#endif

__device__ __forceinline__ v8f wmma16(v16h a, v16h b, v8f c) {
  // D = A(16x32 f16) * B(32x16 f16) + C(16x16 f32)
  return __builtin_amdgcn_wmma_f32_16x16x32_f16(
      /*neg_a=*/false, a, /*neg_b=*/false, b,
      /*c_mod=*/(short)0, c, /*reuse_a=*/false, /*reuse_b=*/false);
}

// A-fragment assembly: lane's 16 halves are two contiguous 16B chunks
// (K = kc + half*16 + 0..7  and  K = kc + half*16 + 8..15).
__device__ __forceinline__ v16h combineA(const _Float16* lo, const _Float16* hi) {
  v8h l = *(const v8h*)lo;
  v8h h = *(const v8h*)hi;
  v16h r;
#pragma unroll
  for (int i = 0; i < 8; ++i) { r[i] = l[i]; r[i + 8] = h[i]; }
  return r;
}

// ---------------------------------------------------------------- pack kernels
__global__ void to_f16(const float* __restrict__ src, _Float16* __restrict__ dst, int n) {
  int i = blockIdx.x * blockDim.x + threadIdx.x;
  if (i < n) dst[i] = (_Float16)src[i];
}

// src: row-major [K][N] f32 ; dst: row-major [N][K] f16  (weightT for B-operand)
__global__ void transpose_to_f16(const float* __restrict__ src, _Float16* __restrict__ dst,
                                 int K, int N) {
  int idx = blockIdx.x * blockDim.x + threadIdx.x;
  if (idx >= K * N) return;
  int k = idx / N, n = idx - k * N;
  dst[(long)n * K + k] = (_Float16)src[idx];
}

// ------------------------------------------------------------- QKV projection
// One wave computes one 16x16 output tile: qkv[M=16384][N=768] = xh @ w_qkvT
__global__ __launch_bounds__(256) void qkv_gemm(
    const _Float16* __restrict__ xh, const _Float16* __restrict__ wT,
    const float* __restrict__ bias,
    _Float16* __restrict__ qh, _Float16* __restrict__ kh, _Float16* __restrict__ vth) {
  const int lane = threadIdx.x & 31;
  const int half = lane >> 4, lcol = lane & 15;
  const int wid  = (blockIdx.x * blockDim.x + threadIdx.x) >> 5;
  const int tn = wid % 48;
  const int tm = wid / 48;

  const _Float16* ap = xh + (long)(tm * 16 + lcol) * DIM + half * 16;   // A row-major
  const _Float16* bp = wT + (long)(tn * 16 + lcol) * DIM + half * 16;  // B column (contig K)
  v8f c = {};
#pragma unroll
  for (int kc = 0; kc < DIM; kc += 32) {
    v16h a = combineA(ap + kc, ap + kc + 8);
    v16h b = *(const v16h*)(bp + kc);
    c = wmma16(a, b, c);
  }
  const int col = tn * 16 + lcol;
  const float bv = bias[col];
#pragma unroll
  for (int r = 0; r < 8; ++r) {
    const int gm = tm * 16 + half * 8 + r;     // C layout: row = r + 8*(lane/16)
    float v = c[r] + bv;
    if (col < 256) {
      qh[(long)gm * DIM + col] = (_Float16)(v * SCALE);      // pre-scale Q
    } else if (col < 512) {
      kh[(long)gm * DIM + (col - 256)] = (_Float16)v;
    } else {
      const int bb = gm >> 11, nn = gm & (NSEQ - 1);
      vth[((long)bb * DIM + (col - 512)) * NSEQ + nn] = (_Float16)v;   // V transposed [b,d,n]
    }
  }
}

// ------------------------------------------------------------------ RoPE fixup
// Rotate channel pairs (2p, 2p+1), p = 0..15, of Q and K in place (f16).
__global__ void rope_inplace(_Float16* __restrict__ qh, _Float16* __restrict__ kh) {
  int idx = blockIdx.x * blockDim.x + threadIdx.x;
  if (idx >= MTOT * 16) return;
  int row = idx >> 4, p = idx & 15;
  int pos = row & (NSEQ - 1);
  float ang = (float)pos * __powf(10000.0f, -(float)p * (1.0f / 16.0f));
  float sn, cs;
  __sincosf(ang, &sn, &cs);
  long base = (long)row * DIM + 2 * p;
  float q0 = (float)qh[base], q1 = (float)qh[base + 1];
  qh[base]     = (_Float16)(q0 * cs - q1 * sn);
  qh[base + 1] = (_Float16)(q1 * cs + q0 * sn);
  float k0 = (float)kh[base], k1 = (float)kh[base + 1];
  kh[base]     = (_Float16)(k0 * cs - k1 * sn);
  kh[base + 1] = (_Float16)(k1 * cs + k0 * sn);
}

// ---------------------------------------------------------------- flash attn
// One wave per (batch, 16-row query tile). Online softmax, O kept in 128 VGPRs.
__global__ __launch_bounds__(32) void flash_attn(
    const _Float16* __restrict__ qh, const _Float16* __restrict__ kh,
    const _Float16* __restrict__ vth, _Float16* __restrict__ ah) {
  const int lane = threadIdx.x & 31;
  const int half = lane >> 4;
  const int lcol = lane & 15;
  const int srow = half * 8;
  const int tile = blockIdx.x;                 // 0..1023
  const int b  = tile >> 7;
  const int i0 = (tile & 127) << 4;
  const long rowbase = (long)b * NSEQ + i0;

  __shared__ __align__(16) _Float16 qs[16 * DIM];   // Q tile (scaled+roped)
  __shared__ __align__(16) float    sf[16 * 32];    // score tile
  __shared__ __align__(16) _Float16 ps[16 * 32];    // prob tile f16
  __shared__ float red[32];
  __shared__ float mrow[16], lrow[16], corr[16];

  // ---- stage Q tile into LDS (async memory->LDS when available)
#ifdef HAVE_ASYNC_LDS
#pragma unroll
  for (int t = 0; t < 16; ++t) {
    int off = t * 256 + lane * 8;
    g_v4i* gsrc = (g_v4i*)(unsigned long long)(qh + rowbase * DIM + off);
    l_v4i* ldst = (l_v4i*)(unsigned int)(unsigned long long)(qs + off);
    __builtin_amdgcn_global_load_async_to_lds_b128(gsrc, ldst, 0, 0);
  }
#else
#pragma unroll
  for (int t = 0; t < 16; ++t) {
    int off = t * 256 + lane * 8;
    *(v8h*)(qs + off) = *(const v8h*)(qh + rowbase * DIM + off);
  }
#endif
  if (lane < 16) { mrow[lane] = -1e30f; lrow[lane] = 0.0f; }

  v8f o[16];
  v8f zero = {};
#pragma unroll
  for (int t = 0; t < 16; ++t) o[t] = zero;

#ifdef HAVE_ASYNC_LDS
#if __has_builtin(__builtin_amdgcn_s_wait_asynccnt)
  __builtin_amdgcn_s_wait_asynccnt(0);
#else
  asm volatile("s_wait_asynccnt 0x0" ::: "memory");
#endif
#endif
  __syncthreads();

  const _Float16* kb = kh  + (long)b * NSEQ * DIM;
  const _Float16* vb = vth + (long)b * DIM * NSEQ;

  for (int j0 = 0; j0 < NSEQ; j0 += 32) {
    // ---- S = Qtile @ K^T  (two 16x16 C-tiles, 16 WMMA)
    v8f s0 = zero, s1 = zero;
    const _Float16* krow0 = kb + (long)(j0 + lcol) * DIM + half * 16;
    const _Float16* krow1 = krow0 + 16 * DIM;
#pragma unroll
    for (int kc = 0; kc < DIM; kc += 32) {
      const _Float16* apq = qs + lcol * DIM + kc + half * 16;   // A-frag from LDS
      v16h a  = combineA(apq, apq + 8);
      v16h b0 = *(const v16h*)(krow0 + kc);
      v16h b1 = *(const v16h*)(krow1 + kc);
      s0 = wmma16(a, b0, s0);
      s1 = wmma16(a, b1, s1);
    }
#pragma unroll
    for (int r = 0; r < 8; ++r) {
      sf[(srow + r) * 32 + lcol]      = s0[r];
      sf[(srow + r) * 32 + 16 + lcol] = s1[r];
    }
    __syncthreads();

    // ---- online softmax: lane owns (row = lcol, cols = half*16 .. +15)
    float mx = -1e30f;
#pragma unroll
    for (int c2 = 0; c2 < 16; ++c2)
      mx = fmaxf(mx, sf[lcol * 32 + half * 16 + c2]);
    red[lane] = mx;
    __syncthreads();
    float rowmax = fmaxf(mx, red[lane ^ 16]);
    float mold = mrow[lcol];
    float mnew = fmaxf(mold, rowmax);
    float csc  = __expf(mold - mnew);
    float psum = 0.0f;
#pragma unroll
    for (int c2 = 0; c2 < 16; ++c2) {
      float e = __expf(sf[lcol * 32 + half * 16 + c2] - mnew);
      ps[lcol * 32 + half * 16 + c2] = (_Float16)e;
      psum += e;
    }
    __syncthreads();
    red[lane] = psum;
    __syncthreads();
    float lnew = csc * lrow[lcol] + psum + red[lane ^ 16];
    if (half == 0) { mrow[lcol] = mnew; lrow[lcol] = lnew; corr[lcol] = csc; }
    __syncthreads();

    // rescale O accumulators by per-row correction
    float cv[8];
#pragma unroll
    for (int r = 0; r < 8; ++r) cv[r] = corr[srow + r];
#pragma unroll
    for (int t = 0; t < 16; ++t)
#pragma unroll
      for (int r = 0; r < 8; ++r) o[t][r] *= cv[r];

    // ---- O += P @ V   (16 WMMA over d-tiles)
    const _Float16* pp = ps + lcol * 32 + half * 16;
    v16h pa = combineA(pp, pp + 8);
    if (j0 + 32 < NSEQ) {
      __builtin_prefetch(krow0 + 32 * DIM, 0, 1);   // -> global_prefetch_b8
      __builtin_prefetch(krow1 + 32 * DIM, 0, 1);
    }
    const _Float16* vcol = vb + j0 + half * 16 + (long)lcol * NSEQ;
#pragma unroll
    for (int t = 0; t < 16; ++t) {
      v16h bv = *(const v16h*)(vcol + (long)t * 16 * NSEQ);
      o[t] = wmma16(pa, bv, o[t]);
    }
    __syncthreads();
  }

  // ---- normalize and store f16 attention output
  if (half == 0) corr[lcol] = 1.0f / lrow[lcol];
  __syncthreads();
  float iv[8];
#pragma unroll
  for (int r = 0; r < 8; ++r) iv[r] = corr[srow + r];
#pragma unroll
  for (int t = 0; t < 16; ++t)
#pragma unroll
    for (int r = 0; r < 8; ++r)
      ah[(rowbase + srow + r) * DIM + t * 16 + lcol] = (_Float16)(o[t][r] * iv[r]);
}

// -------------------------------------------------------------- out projection
__global__ __launch_bounds__(256) void out_gemm(
    const _Float16* __restrict__ ahm, const _Float16* __restrict__ wT,
    const float* __restrict__ bias, float* __restrict__ out) {
  const int lane = threadIdx.x & 31;
  const int half = lane >> 4, lcol = lane & 15;
  const int wid  = (blockIdx.x * blockDim.x + threadIdx.x) >> 5;
  const int tn = wid & 15;
  const int tm = wid >> 4;

  const _Float16* ap = ahm + (long)(tm * 16 + lcol) * DIM + half * 16;
  const _Float16* bp = wT  + (long)(tn * 16 + lcol) * DIM + half * 16;
  v8f c = {};
#pragma unroll
  for (int kc = 0; kc < DIM; kc += 32) {
    v16h a = combineA(ap + kc, ap + kc + 8);
    v16h b = *(const v16h*)(bp + kc);
    c = wmma16(a, b, c);
  }
  const int col = tn * 16 + lcol;
  const float bv = bias[col];
#pragma unroll
  for (int r = 0; r < 8; ++r) {
    const int gm = tm * 16 + half * 8 + r;
    out[(long)gm * DIM + col] = c[r] + bv;
  }
}

// ------------------------------------------------------------------- launcher
extern "C" void kernel_launch(void* const* d_in, const int* in_sizes, int n_in,
                              void* d_out, int out_size, void* d_ws, size_t ws_size,
                              hipStream_t stream) {
  const float* x     = (const float*)d_in[0];
  const float* w_qkv = (const float*)d_in[1];
  const float* b_qkv = (const float*)d_in[2];
  const float* w_out = (const float*)d_in[3];
  const float* b_out = (const float*)d_in[4];
  float* out = (float*)d_out;

  // workspace layout (f16), total ~42.5 MB
  _Float16* xh  = (_Float16*)d_ws;
  _Float16* wqT = xh  + (long)MTOT * DIM;        // [768][256]
  _Float16* woT = wqT + 768 * 256;               // [256][256]
  _Float16* qh  = woT + 256 * 256;               // [16384][256] scaled+roped
  _Float16* kh  = qh  + (long)MTOT * DIM;        // [16384][256] roped
  _Float16* vth = kh  + (long)MTOT * DIM;        // [8][256][2048] V transposed
  _Float16* ah  = vth + (long)MTOT * DIM;        // [16384][256] attn out

  const int nx = MTOT * DIM;
  to_f16<<<(nx + 255) / 256, 256, 0, stream>>>(x, xh, nx);
  transpose_to_f16<<<(256 * 768 + 255) / 256, 256, 0, stream>>>(w_qkv, wqT, 256, 768);
  transpose_to_f16<<<(256 * 256 + 255) / 256, 256, 0, stream>>>(w_out, woT, 256, 256);
  qkv_gemm<<<(1024 * 48) / 8, 256, 0, stream>>>(xh, wqT, b_qkv, qh, kh, vth);
  rope_inplace<<<(MTOT * 16 + 255) / 256, 256, 0, stream>>>(qh, kh);
  flash_attn<<<NB * (NSEQ / 16), 32, 0, stream>>>(qh, kh, vth, ah);
  out_gemm<<<(1024 * 16) / 8, 256, 0, stream>>>(ah, woT, b_out, out);
}
